// DiscoveryNet_51539608363
// MI455X (gfx1250) — matrix operand order repaired and checked by reading, not compiled, not run on hardware
//
#include <hip/hip_runtime.h>

typedef __attribute__((ext_vector_type(16))) _Float16 v16h;
typedef __attribute__((ext_vector_type(8)))  float    v8f;

#define B_ 512
#define N_ 64
#define H_ 128

// tanh(x) = 1 - 2/(exp(2x)+1), on the fast transcendental pipe (v_exp_f32, v_rcp_f32)
__device__ __forceinline__ float fast_tanh(float x) {
  float e = __builtin_amdgcn_exp2f(x * 2.8853900817779268f); // 2*log2(e)
  return 1.0f - 2.0f * __builtin_amdgcn_rcpf(e + 1.0f);
}

// upper-triangle coarse-tile pairs (16x16 atom blocks), ct_i <= ct_j
__device__ __constant__ signed char kCtI[10] = {0,0,0,0,1,1,1,2,2,3};
__device__ __constant__ signed char kCtJ[10] = {0,1,2,3,1,2,3,2,3,3};

__global__ __launch_bounds__(256)
void force_mlp_kernel(const float* __restrict__ pos,
                      const float* __restrict__ W1, const float* __restrict__ b1,
                      const float* __restrict__ W2, const float* __restrict__ b2,
                      const float* __restrict__ W3, const float* __restrict__ b3,
                      float* __restrict__ out)
{
  __shared__ __align__(32) _Float16 sW2t[H_ * H_]; // W2 transposed: [n_out][k], f16
  __shared__ float sW1[6 * H_];
  __shared__ float sB1[H_];
  __shared__ float sB2[H_];
  __shared__ float sW3[H_];
  __shared__ float sPos[N_ * 3];
  __shared__ float sMag[N_ * N_];   // [i][j], filled only where (i>>4)<=(j>>4)
  __shared__ float sF[N_ * 3];      // per-i force accumulators

  const int tid = threadIdx.x;
  const int b   = blockIdx.x;

  // ---- stage weights / positions into LDS ----
  for (int idx = tid; idx < H_ * H_; idx += 256) {
    int c = idx >> 7, k = idx & 127;                 // sW2t[c][k] = W2[k][c]
    sW2t[idx] = (_Float16)W2[k * H_ + c];
  }
  for (int idx = tid; idx < 6 * H_; idx += 256) sW1[idx] = W1[idx];
  if (tid < H_) { sB1[tid] = b1[tid]; sB2[tid] = b2[tid]; sW3[tid] = W3[tid]; }
  if (tid < N_ * 3) { sPos[tid] = pos[b * N_ * 3 + tid]; sF[tid] = 0.0f; }
  __syncthreads();

  const int wv    = tid >> 5;      // wave id 0..7
  const int lane  = tid & 31;
  const int lhalf = lane >> 4;     // 0 or 1 (K half in A/B fragment layout)
  const int lrow  = lane & 15;     // row within M-tile / column within N-tile
  const float b3v = b3[0];

  // ---- 160 M-tiles per block (upper triangle only), 20 per wave ----
  #pragma unroll 1
  for (int tt = 0; tt < 20; ++tt) {
    const int t  = wv * 20 + tt;        // 0..159
    const int cp = t >> 4;              // coarse pair 0..9
    const int s  = t & 15;              // i within coarse pair
    const int i  = (int)kCtI[cp] * 16 + s;
    const int jc = (int)kCtJ[cp] * 16;  // j-chunk base
    const int j  = jc + lrow;

    // features for pair (i, j) -- mag is symmetric in (i,j)
    float dx = sPos[i*3+0] - sPos[j*3+0];
    float dy = sPos[i*3+1] - sPos[j*3+1];
    float dz = sPos[i*3+2] - sPos[j*3+2];
    float sq = dx*dx + dy*dy + dz*dz;
    float dist = sq > 0.0f ? sqrtf(sq) : 0.0f;
    float inv_r = 1.0f / fmaxf(dist, 0.5f);
    float i2 = inv_r * inv_r;
    float inv6  = i2 * i2 * i2;
    float inv12 = inv6 * inv6;
    const float f0 = dist, f1 = inv_r, f2 = inv6, f3 = inv12,
                f4 = inv6 * inv_r, f5 = inv12 * inv_r;

    // layer 1: each lane computes exactly the 64 h1 columns it must hold
    // in the 16x32 f16 A-fragment layout (4 chunks of K=32)
    v16h afrag[4];
    #pragma unroll
    for (int kc = 0; kc < 4; ++kc) {
      #pragma unroll
      for (int e = 0; e < 16; ++e) {
        int c = kc * 32 + lhalf * 16 + e;
        float h = sB1[c];
        h = fmaf(f0, sW1[0*H_ + c], h);
        h = fmaf(f1, sW1[1*H_ + c], h);
        h = fmaf(f2, sW1[2*H_ + c], h);
        h = fmaf(f3, sW1[3*H_ + c], h);
        h = fmaf(f4, sW1[4*H_ + c], h);
        h = fmaf(f5, sW1[5*H_ + c], h);
        afrag[kc][e] = (_Float16)fast_tanh(h);
      }
    }

    // layer 2 + layer 3 fused: two v8f accumulators live at a time
    // (two independent WMMA chains, no spills)
    float p[8];
    #pragma unroll
    for (int r = 0; r < 8; ++r) p[r] = 0.0f;

    #pragma unroll 1
    for (int n = 0; n < 8; n += 2) {
      v8f acc0 = {};
      v8f acc1 = {};
      #pragma unroll
      for (int kc = 0; kc < 4; ++kc) {
        const v16h bf0 =
            *(const v16h*)(&sW2t[((n + 0) * 16 + lrow) * H_ + kc * 32 + lhalf * 16]);
        const v16h bf1 =
            *(const v16h*)(&sW2t[((n + 1) * 16 + lrow) * H_ + kc * 32 + lhalf * 16]);
        acc0 = __builtin_amdgcn_wmma_f32_16x16x32_f16(
            false, afrag[kc], false, bf0, (short)0, acc0, false, false);
        acc1 = __builtin_amdgcn_wmma_f32_16x16x32_f16(
            false, afrag[kc], false, bf1, (short)0, acc1, false, false);
      }
      // bias + tanh + W3 partial dot (C/D layout: VGPR r <-> row, lane <-> col)
      const int c0 = (n + 0) * 16 + lrow;
      const int c1 = (n + 1) * 16 + lrow;
      const float bb0 = sB2[c0], w30 = sW3[c0];
      const float bb1 = sB2[c1], w31 = sW3[c1];
      #pragma unroll
      for (int r = 0; r < 8; ++r) {
        p[r] = fmaf(fast_tanh(acc0[r] + bb0), w30, p[r]);
        p[r] = fmaf(fast_tanh(acc1[r] + bb1), w31, p[r]);
      }
    }

    // reduce over the 16 lanes that share each row group
    #pragma unroll
    for (int m = 1; m < 16; m <<= 1) {
      #pragma unroll
      for (int r = 0; r < 8; ++r) p[r] += __shfl_xor(p[r], m, 32);
    }
    if (lrow == 0) {
      int colbase = jc + lhalf * 8;   // lanes 0 and 16 cover j-cols 0-7 / 8-15
      #pragma unroll
      for (int r = 0; r < 8; ++r)
        sMag[i * N_ + colbase + r] = p[r] + b3v;
    }
  }
  __syncthreads();

  // ---- masked direction-weighted reduction over j (use symmetry for j<i tiles) ----
  {
    const int i  = tid >> 2;          // 0..63
    const int j0 = (tid & 3) * 16;    // 4 threads per i, 16 j's each
    float fx = 0.0f, fy = 0.0f, fz = 0.0f;
    #pragma unroll 1
    for (int jj = 0; jj < 16; ++jj) {
      int j = j0 + jj;
      if (j != i) {
        float dx = sPos[i*3+0] - sPos[j*3+0];
        float dy = sPos[i*3+1] - sPos[j*3+1];
        float dz = sPos[i*3+2] - sPos[j*3+2];
        float sq = dx*dx + dy*dy + dz*dz;
        float dist = sq > 0.0f ? sqrtf(sq) : 0.0f;
        float inv = 1.0f / fmaxf(dist, 0.01f);
        // mag is symmetric; only (i>>4) <= (j>>4) entries were computed
        float m = ((i >> 4) <= (j >> 4)) ? sMag[i * N_ + j] : sMag[j * N_ + i];
        m *= inv;
        fx = fmaf(m, dx, fx);
        fy = fmaf(m, dy, fy);
        fz = fmaf(m, dz, fz);
      }
    }
    atomicAdd(&sF[i*3 + 0], fx);   // ds_add_f32
    atomicAdd(&sF[i*3 + 1], fy);
    atomicAdd(&sF[i*3 + 2], fz);
  }
  __syncthreads();

  if (tid < N_ * 3) {
    out[b * N_ * 3 + tid] = sF[tid];
  }
}

extern "C" void kernel_launch(void* const* d_in, const int* in_sizes, int n_in,
                              void* d_out, int out_size, void* d_ws, size_t ws_size,
                              hipStream_t stream) {
  (void)in_sizes; (void)n_in; (void)out_size; (void)d_ws; (void)ws_size;
  const float* pos = (const float*)d_in[0];
  const float* W1  = (const float*)d_in[1];
  const float* b1  = (const float*)d_in[2];
  const float* W2  = (const float*)d_in[3];
  const float* b2  = (const float*)d_in[4];
  const float* W3  = (const float*)d_in[5];
  const float* b3  = (const float*)d_in[6];
  float* out = (float*)d_out;

  dim3 grid(B_);       // one block per batch element
  dim3 block(256);     // 8 wave32
  force_mlp_kernel<<<grid, block, 0, stream>>>(pos, W1, b1, W2, b2, W3, b3, out);
}
